// SpanEncoder_36661840839550
// MI455X (gfx1250) — matrix operand order repaired and checked by reading, not compiled, not run on hardware
//
#include <hip/hip_runtime.h>
#include <hip/hip_bf16.h>

// ---------- types ----------
typedef __bf16 v16bf __attribute__((ext_vector_type(16)));
typedef __bf16 v8bf  __attribute__((ext_vector_type(8)));
typedef float  v8f   __attribute__((ext_vector_type(8)));
typedef float  f4    __attribute__((ext_vector_type(4)));

#define HDIM 512
#define LLEN 128
#define BSZ  4
#define TSTEPS 130          // L + 2
#define G4   2048           // 4*H
#define MROWS 520           // TSTEPS*BSZ
#define MPAD  528           // padded to 33 * 16
#define NWG_DIR 16u         // recurrence workgroups per direction

// =====================================================================
// prep: f32->bf16 weight conversion, padded x in bf16, zero state/sync
// =====================================================================
__global__ __launch_bounds__(256) void prep_kernel(
    const float* __restrict__ wr,
    const float* __restrict__ Wihf, const float* __restrict__ Whhf,
    const float* __restrict__ Wihb, const float* __restrict__ Whhb,
    __bf16* __restrict__ wbf, __bf16* __restrict__ xbf,
    __bf16* __restrict__ h16, float* __restrict__ cbuf,
    unsigned* __restrict__ sync)
{
    size_t tid = (size_t)blockIdx.x * blockDim.x + threadIdx.x;
    size_t nth = (size_t)gridDim.x * blockDim.x;
    const size_t WN = (size_t)G4 * HDIM;   // 1M elements per matrix

    // weights -> bf16 : [Wih_f][Whh_f][Wih_b][Whh_b]
    for (size_t i = tid; i < 4 * WN; i += nth) {
        size_t w = i / WN, r = i % WN;
        const float* src = (w == 0) ? Wihf : (w == 1) ? Whhf : (w == 2) ? Wihb : Whhb;
        wbf[i] = (__bf16)src[r];
    }
    // padded x: row = t*4+b, t in [0,130); t==0, t==129, rows>=520 are zero
    for (size_t i = tid; i < (size_t)MPAD * HDIM; i += nth) {
        size_t row = i / HDIM, c = i % HDIM;
        size_t t = row / BSZ, b = row % BSZ;
        float v = 0.f;
        if (row < MROWS && t >= 1 && t <= LLEN)
            v = wr[(b * (size_t)LLEN + (t - 1)) * HDIM + c];
        xbf[i] = (__bf16)v;
    }
    // h double buffers (2 dirs x 2 bufs x 16x512), c state, sync counters
    for (size_t i = tid; i < (size_t)2 * 2 * 16 * HDIM; i += nth) h16[i] = (__bf16)0.f;
    for (size_t i = tid; i < (size_t)2 * BSZ * HDIM; i += nth)    cbuf[i] = 0.f;
    if (tid < 128) sync[tid] = 0u;
}

// =====================================================================
// GX[d][t*4+b][n] = x_pad[t][b] @ W_ih_d^T + bias_d     (bf16 WMMA)
// one wave per 16x64 tile, K=512
// =====================================================================
__global__ __launch_bounds__(32) void gemm_gx_kernel(
    const __bf16* __restrict__ wbf, const __bf16* __restrict__ xbf,
    const float* __restrict__ bias_f, const float* __restrict__ bias_b,
    float* __restrict__ GX)
{
    const int lane = threadIdx.x;
    const int ng   = blockIdx.x;          // 0..31 (64-col group)
    const int mt   = blockIdx.y;          // 0..32
    const int dir  = blockIdx.z;          // 0..1

    const __bf16* Wih  = wbf + (size_t)dir * 2 * (size_t)G4 * HDIM; // slot 0 / 2
    const float*  bias = dir ? bias_b : bias_f;

    const int mrow   = mt * 16 + (lane & 15);           // A: M = lane%16
    const int khalfA = (lane < 16) ? 0 : 16;            // A K split: 0-15 vs 16-31
    const int khalfB = (lane < 16) ? 0 : 16;            // B K split: 0-15 vs 16-31
    const __bf16* arow = xbf + (size_t)mrow * HDIM;
    const int n0 = ng * 64;

    v8f acc[4] = {};
#pragma unroll
    for (int kc = 0; kc < 16; ++kc) {
        const int kb = kc * 32;
        v8bf alo = *(const v8bf*)(arow + kb + khalfA);
        v8bf ahi = *(const v8bf*)(arow + kb + khalfA + 8);
        v16bf a;
#pragma unroll
        for (int i = 0; i < 8; ++i) { a[i] = alo[i]; a[8 + i] = ahi[i]; }
#pragma unroll
        for (int nt = 0; nt < 4; ++nt) {
            const int n = n0 + nt * 16 + (lane & 15);   // B: N = lane%16
            v16bf b = *(const v16bf*)(Wih + (size_t)n * HDIM + kb + khalfB);
            acc[nt] = __builtin_amdgcn_wmma_f32_16x16x32_bf16(
                false, a, false, b, (short)0, acc[nt], false, false);
        }
    }
    // C/D layout: lane col = n0+nt*16+lane%16 ; VGPR r -> row r + (lane<16?0:8)
    const int rbase = (lane < 16) ? 0 : 8;
    float* GXd = GX + (size_t)dir * MROWS * G4;
#pragma unroll
    for (int nt = 0; nt < 4; ++nt) {
        const int col = n0 + nt * 16 + (lane & 15);
        const float bv = bias[col];
#pragma unroll
        for (int r = 0; r < 8; ++r) {
            const int row = mt * 16 + rbase + r;
            if (row < MROWS)
                GXd[(size_t)row * G4 + col] = acc[nt][r] + bv;
        }
    }
}

// =====================================================================
// per-direction barrier: single monotonic counter, no reset, no winner
// =====================================================================
__device__ __forceinline__ void dir_barrier(unsigned* cnt, unsigned target, int tid)
{
    __syncthreads();
    if (tid == 0) {
        __threadfence();
        __atomic_fetch_add(cnt, 1u, __ATOMIC_RELEASE);
        while (__atomic_load_n(cnt, __ATOMIC_ACQUIRE) < target)
            __builtin_amdgcn_s_sleep(1);
        __threadfence();
    }
    __syncthreads();
}

// =====================================================================
// persistent BiLSTM recurrence: W_hh register-resident, 130 steps.
// grid = 32 WGs: dir (2) x 16 WGs; 8 waves/WG = 2 j-tiles x 4 gates.
// Directions are fully decoupled (independent barrier counters).
// =====================================================================
__global__ __launch_bounds__(256) void lstm_recurrence_kernel(
    const __bf16* __restrict__ wbf, const float* __restrict__ GX,
    __bf16* __restrict__ h16, float* __restrict__ cbuf,
    float* __restrict__ Hf, float* __restrict__ Hb,
    unsigned* __restrict__ sync)
{
    const int tid  = threadIdx.x;
    const int wave = tid >> 5;
    const int lane = tid & 31;
    const int gate = wave & 3;        // i,f,g,o
    const int tile = wave >> 2;       // 0..1  (j-tile within WG)
    const int wg   = blockIdx.x;      // 0..31
    const int dir  = wg >> 4;
    const int jt   = (wg & 15) * 2 + tile;     // 0..31

    const __bf16* Whh = wbf + ((size_t)dir * 2 + 1) * (size_t)G4 * HDIM;

    // preload B operand: W_hh rows n = gate*512 + jt*16 + lane%16, K=512
    const int n      = gate * HDIM + jt * 16 + (lane & 15);
    const int khalfB = (lane < 16) ? 0 : 16;
    const __bf16* wrow = Whh + (size_t)n * HDIM;
    v16bf bw[16];
#pragma unroll
    for (int kc = 0; kc < 16; ++kc)
        bw[kc] = *(const v16bf*)(wrow + kc * 32 + khalfB);

    __bf16* h16d = h16 + (size_t)dir * 2 * (16 * HDIM);
    float*  cd   = cbuf + (size_t)dir * BSZ * HDIM;
    const float* GXd = GX + (size_t)dir * MROWS * G4;
    float* Hd = dir ? Hb : Hf;
    unsigned* cnt = sync + (size_t)dir * 64;   // 256B apart

    const int m      = lane & 15;
    const int khalfA = (lane < 16) ? 0 : 16;
    const int col    = gate * HDIM + jt * 16 + lane;  // valid when lane<16

    __shared__ float gshare[2][4][4][16];   // [tile][gate][batch][j-local]

    for (int t = 0; t < TSTEPS; ++t) {
        const int cur = t & 1;
        const __bf16* hrow = h16d + cur * (16 * HDIM) + (size_t)m * HDIM;

        // 4 independent accumulator chains (kc mod 4) -> 4x shorter critical path
        v8f accs[4] = {};
#pragma unroll
        for (int kc = 0; kc < 16; ++kc) {
            const int kb = kc * 32;
            v8bf alo = *(const v8bf*)(hrow + kb + khalfA);
            v8bf ahi = *(const v8bf*)(hrow + kb + khalfA + 8);
            v16bf a;
#pragma unroll
            for (int i = 0; i < 8; ++i) { a[i] = alo[i]; a[8 + i] = ahi[i]; }
            accs[kc & 3] = __builtin_amdgcn_wmma_f32_16x16x32_bf16(
                false, a, false, bw[kc], (short)0, accs[kc & 3], false, false);
        }
        v8f acc = (accs[0] + accs[1]) + (accs[2] + accs[3]);

        const int xt = dir ? (TSTEPS - 1 - t) : t;   // xs index (backward reversed)
        if (lane < 16) {
#pragma unroll
            for (int r = 0; r < 4; ++r) {            // only batch rows 0..3 valid
                const int row = xt * BSZ + r;
                gshare[tile][gate][r][lane] = acc[r] + GXd[(size_t)row * G4 + col];
            }
            // prefetch next step's GX rows (data already resident from gemm)
            if (t + 1 < TSTEPS) {
                const int xtn = dir ? (TSTEPS - 2 - t) : (t + 1);
#pragma unroll
                for (int r = 0; r < 4; ++r)
                    __builtin_prefetch(&GXd[(size_t)(xtn * BSZ + r) * G4 + col], 0, 3);
            }
        }
        __syncthreads();

        if (tid < 128) {
            const int tl = tid >> 6;                 // j-tile 0..1
            const int mm = (tid >> 4) & 3;           // batch
            const int jl = tid & 15;
            const int j  = ((wg & 15) * 2 + tl) * 16 + jl;
            const float gi = gshare[tl][0][mm][jl];
            const float gf = gshare[tl][1][mm][jl];
            const float gg = gshare[tl][2][mm][jl];
            const float go = gshare[tl][3][mm][jl];
            const float c_old = cd[mm * HDIM + j];
            const float si = 1.f / (1.f + __expf(-gi));
            const float sf = 1.f / (1.f + __expf(-gf));
            const float so = 1.f / (1.f + __expf(-go));
            const float c_new = sf * c_old + si * tanhf(gg);
            const float h     = so * tanhf(c_new);
            cd[mm * HDIM + j] = c_new;
            h16d[(cur ^ 1) * (16 * HDIM) + mm * HDIM + j] = (__bf16)h;
            Hd[((size_t)xt * BSZ + mm) * HDIM + j] = h;   // natural-time history
        }

        dir_barrier(cnt, NWG_DIR * (unsigned)(t + 1), tid);
    }
}

// =====================================================================
// out[i][k][b][0:512]    = Hf[k][b] - Hf[i][b]
// out[i][k][b][512:1024] = Hb[i+1][b] - Hb[k+1][b]
// masked by (i>=1 && k>i && k<=126); fully coalesced NT b128 stores
// =====================================================================
__global__ __launch_bounds__(256) void span_output_kernel(
    const float* __restrict__ Hf, const float* __restrict__ Hb,
    float* __restrict__ out)
{
    const int k = blockIdx.x;
    const int i = blockIdx.y;
    const int tid = threadIdx.x;
    const bool valid = (i >= 1) && (k > i) && (k <= LLEN - 2);

    float* base = out + (((size_t)i * LLEN + k) * BSZ) * 1024;

    if (!valid) {
        f4 z = {0.f, 0.f, 0.f, 0.f};
#pragma unroll
        for (int b = 0; b < 4; ++b)
            __builtin_nontemporal_store(z, (f4*)(base + (size_t)b * 1024 + tid * 4));
        return;
    }
    if (tid < 128) {                       // first half: forward features
        const int c = tid * 4;
#pragma unroll
        for (int b = 0; b < 4; ++b) {
            f4 r = *(const f4*)(Hf + ((size_t)k * BSZ + b) * HDIM + c)
                 - *(const f4*)(Hf + ((size_t)i * BSZ + b) * HDIM + c);
            __builtin_nontemporal_store(r, (f4*)(base + (size_t)b * 1024 + c));
        }
    } else {                               // second half: backward features
        const int cc = tid * 4 - HDIM;
#pragma unroll
        for (int b = 0; b < 4; ++b) {
            f4 r = *(const f4*)(Hb + ((size_t)(i + 1) * BSZ + b) * HDIM + cc)
                 - *(const f4*)(Hb + ((size_t)(k + 1) * BSZ + b) * HDIM + cc);
            __builtin_nontemporal_store(r, (f4*)(base + (size_t)b * 1024 + HDIM + cc));
        }
    }
}

// =====================================================================
extern "C" void kernel_launch(void* const* d_in, const int* in_sizes, int n_in,
                              void* d_out, int out_size, void* d_ws, size_t ws_size,
                              hipStream_t stream)
{
    const float* wr   = (const float*)d_in[0];
    const float* Wihf = (const float*)d_in[1];
    const float* Whhf = (const float*)d_in[2];
    const float* bf   = (const float*)d_in[3];
    const float* Wihb = (const float*)d_in[4];
    const float* Whhb = (const float*)d_in[5];
    const float* bb   = (const float*)d_in[6];

    char* ws = (char*)d_ws;
    size_t off = 0;
    auto take = [&](size_t bytes) {
        size_t o = off;
        off = (off + bytes + 255) & ~(size_t)255;
        return o;
    };
    unsigned* sync = (unsigned*)(ws + take(512));
    __bf16*   wbf  = (__bf16*)  (ws + take((size_t)4 * G4 * HDIM * 2));      // 8 MB
    __bf16*   xbf  = (__bf16*)  (ws + take((size_t)MPAD * HDIM * 2));        // 0.5 MB
    float*    GX   = (float*)   (ws + take((size_t)2 * MROWS * G4 * 4));     // 8.5 MB
    __bf16*   h16  = (__bf16*)  (ws + take((size_t)2 * 2 * 16 * HDIM * 2));  // 64 KB
    float*    cbuf = (float*)   (ws + take((size_t)2 * BSZ * HDIM * 4));     // 16 KB
    float*    Hf   = (float*)   (ws + take((size_t)TSTEPS * BSZ * HDIM * 4));// 1 MB
    float*    Hb   = (float*)   (ws + take((size_t)TSTEPS * BSZ * HDIM * 4));// 1 MB
    (void)in_sizes; (void)n_in; (void)out_size; (void)ws_size;

    hipLaunchKernelGGL(prep_kernel, dim3(512), dim3(256), 0, stream,
                       wr, Wihf, Whhf, Wihb, Whhb, wbf, xbf, h16, cbuf, sync);
    hipLaunchKernelGGL(gemm_gx_kernel, dim3(32, 33, 2), dim3(32), 0, stream,
                       wbf, xbf, bf, bb, GX);
    hipLaunchKernelGGL(lstm_recurrence_kernel, dim3(2 * NWG_DIR), dim3(256), 0, stream,
                       wbf, GX, h16, cbuf, Hf, Hb, sync);
    hipLaunchKernelGGL(span_output_kernel, dim3(LLEN, LLEN), dim3(256), 0, stream,
                       Hf, Hb, (float*)d_out);
}
